// LeNet_21311627723193
// MI455X (gfx1250) — matrix operand order repaired and checked by reading, not compiled, not run on hardware
//
#include <hip/hip_runtime.h>
#include <hip/hip_bf16.h>

typedef _Float16 h16;
typedef __attribute__((ext_vector_type(16))) _Float16 v16h;
typedef __attribute__((ext_vector_type(8)))  _Float16 v8h;
typedef __attribute__((ext_vector_type(8)))  float    v8f;

#define EPSQ  1e-8f
#define BATCH 16384
#define NEGBIG (-3.0e38f)

// ---- monotonic float<->uint key for atomic max (order-independent => deterministic) ----
__device__ __forceinline__ unsigned fkey(float f){
  int b = __float_as_int(f);
  return (b >= 0) ? ((unsigned)b | 0x80000000u) : ~((unsigned)b);
}
__device__ __forceinline__ float fdecode(unsigned k){
  int b = (k & 0x80000000u) ? (int)(k & 0x7fffffffu) : (int)~k;
  return __int_as_float(b);
}
// int8 fake-quant: clip(round(x/s),-128,127)*s   (round-to-nearest-even like jnp.round)
__device__ __forceinline__ float fq8(float x, float s){
  float q = rintf(x / s);
  q = fminf(fmaxf(q, -128.f), 127.f);
  return q * s;
}
__device__ __forceinline__ float qrelu_f(float x, float sa, float sb){
  float q = fq8(x, sa);
  q = fmaxf(q, 0.f);
  return fq8(q, sb);
}

// ---------------- init: reset reduction keys each call ----------------
__global__ void init_kernel(unsigned* red, float* s){
  int t = threadIdx.x;
  if (t < 16) red[t] = 0u;
  if (t >= 16 && t < 32) s[t - 16] = 1.0f;
}

// ---------------- weight prep: binarize (+/- mean|w|), pad, fp16 ----------------
__global__ void prep_weights_kernel(
    const float* __restrict__ w1, const float* __restrict__ b1,
    const float* __restrict__ w2, const float* __restrict__ b2,
    const float* __restrict__ fw1, const float* __restrict__ fb1,
    const float* __restrict__ fw2, const float* __restrict__ fb2,
    const float* __restrict__ fw3, const float* __restrict__ fb3,
    float* wq1, float* bias1, h16* w2p, float* bias2,
    h16* f1p, float* fb1p, h16* f2p, float* fb2p, h16* f3p, float* fb3p)
{
  __shared__ float sred[256];
  int t = threadIdx.x;
  int which = blockIdx.x;
  const float* w = w1; int cnt = 450;
  if (which == 1){ w = w2;  cnt = 2400;  }
  if (which == 2){ w = fw1; cnt = 48000; }
  if (which == 3){ w = fw2; cnt = 10080; }
  if (which == 4){ w = fw3; cnt = 840;   }
  float local = 0.f;
  for (int i = t; i < cnt; i += 256) local += fabsf(w[i]);
  sred[t] = local; __syncthreads();
  for (int off = 128; off > 0; off >>= 1){
    if (t < off) sred[t] += sred[t + off];
    __syncthreads();
  }
  float sc = sred[0] / (float)cnt;   // deterministic tree reduction

  if (which == 0){
    for (int i = t; i < 450; i += 256) wq1[i] = (w1[i] >= 0.f) ? sc : -sc;
    if (t < 6) bias1[t] = b1[t];
  } else if (which == 1){
    // conv2 weights remapped to GEMM K = dy*32 + dx*6 + ci  (per-row pad 30->32)
    for (int i = t; i < 16*160; i += 256){
      int n = i / 160, kw = i % 160;
      int dy = kw / 32, r = kw % 32;
      float v = 0.f;
      if (r < 30){
        int dx = r / 6, ci = r % 6;
        v = (w2[((n*6 + ci)*5 + dy)*5 + dx] >= 0.f) ? sc : -sc;
      }
      w2p[i] = (h16)v;
    }
    if (t < 16) bias2[t] = b2[t];
  } else if (which == 2){            // [128][416], pad 120->128, 400->416
    for (int i = t; i < 128*416; i += 256){
      int n = i / 416, k = i % 416;
      float v = 0.f;
      if (n < 120 && k < 400) v = (fw1[n*400 + k] >= 0.f) ? sc : -sc;
      f1p[i] = (h16)v;
    }
    if (t < 128) fb1p[t] = (t < 120) ? fb1[t] : 0.f;
  } else if (which == 3){            // [96][128], pad 84->96, 120->128
    for (int i = t; i < 96*128; i += 256){
      int n = i / 128, k = i % 128;
      float v = 0.f;
      if (n < 84 && k < 120) v = (fw2[n*120 + k] >= 0.f) ? sc : -sc;
      f2p[i] = (h16)v;
    }
    if (t < 96) fb2p[t] = (t < 84) ? fb2[t] : 0.f;
  } else {                           // [16][96], pad 10->16, 84->96
    for (int i = t; i < 16*96; i += 256){
      int n = i / 96, k = i % 96;
      float v = 0.f;
      if (n < 10 && k < 84) v = (fw3[n*84 + k] >= 0.f) ? sc : -sc;
      f3p[i] = (h16)v;
    }
    if (t < 16) fb3p[t] = (t < 10) ? fb3[t] : 0.f;
  }
}

// ---------------- global max|x| over input (float4 vectorized) ----------------
__global__ void absmax4_kernel(const float4* __restrict__ x, long n4, unsigned* red){
  __shared__ unsigned sA;
  if (threadIdx.x == 0) sA = 0u;
  __syncthreads();
  long i = (long)blockIdx.x * blockDim.x + threadIdx.x;
  long stride = (long)gridDim.x * blockDim.x;
  float m = 0.f;
  for (; i < n4; i += stride){
    float4 v = x[i];
    m = fmaxf(m, fmaxf(fmaxf(fabsf(v.x), fabsf(v.y)), fmaxf(fabsf(v.z), fabsf(v.w))));
  }
  atomicMax(&sA, fkey(m));
  __syncthreads();
  if (threadIdx.x == 0) atomicMax(red, sA);
}

// ---------------- scale compute (1 thread) ----------------
__global__ void scale0_kernel(const unsigned* red, float* s){
  if (threadIdx.x == 0) s[0] = fmaxf(fdecode(red[0]) / 127.f, EPSQ);
}
// inner scale from max|y|; outer scale via monotonicity: max(relu(fq(y))) = max(fq(max y),0)
__global__ void scale2_kernel(const unsigned* red, float* s, int ra, int rm, int oa, int ob){
  if (threadIdx.x == 0){
    float sa = fmaxf(fdecode(red[ra]) / 127.f, EPSQ);
    float my = fdecode(red[rm]);
    float m2 = fmaxf(fq8(my, sa), 0.f);
    s[oa] = sa;
    s[ob] = fmaxf(m2 / 127.f, EPSQ);
  }
}

// ---------------- conv1 (3->6, 5x5) fused with 2x2 pool + max reductions ----------------
// one thread = one pooled pixel, all 6 channels; writes pooled PRE-qrelu values
// to NHWC fp16: p1[((b*14+y)*14+x)*6 + c]  (6 contiguous halves per thread)
__global__ void __launch_bounds__(256) conv1_pool_kernel(
    const float* __restrict__ in, const float* __restrict__ wq1,
    const float* __restrict__ bias1, const float* __restrict__ s,
    h16* __restrict__ p1, unsigned* red)
{
  __shared__ float lw[456];
  __shared__ unsigned sA, sM;
  int t = threadIdx.x;
  for (int i = t; i < 450; i += 256) lw[i] = wq1[i];
  if (t < 6) lw[450 + t] = bias1[t];
  if (t == 0){ sA = 0u; sM = 0u; }
  __syncthreads();
  float s0 = s[0];
  int idx = blockIdx.x * 256 + t;
  float labs = 0.f, lmax = NEGBIG;
  if (idx < BATCH * 196){
    int b = idx / 196; int r = idx % 196;
    int py = r / 14, px = r % 14;
    int iy0 = py * 2, ix0 = px * 2;
    const float* inb = in + (long)b * 3072;
    float pmax[6];
    #pragma unroll
    for (int o = 0; o < 6; ++o) pmax[o] = NEGBIG;
    #pragma unroll
    for (int dy = 0; dy < 2; ++dy){
      #pragma unroll
      for (int dx = 0; dx < 2; ++dx){
        float acc[6];
        #pragma unroll
        for (int o = 0; o < 6; ++o) acc[o] = lw[450 + o];
        for (int ci = 0; ci < 3; ++ci){
          const float* inc = inb + ci * 1024 + (iy0 + dy) * 32 + (ix0 + dx);
          #pragma unroll
          for (int u = 0; u < 5; ++u){
            #pragma unroll
            for (int v = 0; v < 5; ++v){
              float xq = fq8(inc[u*32 + v], s0);
              const float* wr = &lw[ci*25 + u*5 + v];
              #pragma unroll
              for (int o = 0; o < 6; ++o) acc[o] += xq * wr[o*75];
            }
          }
        }
        #pragma unroll
        for (int o = 0; o < 6; ++o){
          labs = fmaxf(labs, fabsf(acc[o]));
          lmax = fmaxf(lmax, acc[o]);
          pmax[o] = fmaxf(pmax[o], acc[o]);
        }
      }
    }
    h16* po = p1 + (long)idx * 6;          // NHWC: idx == (b*14+py)*14+px
    #pragma unroll
    for (int o = 0; o < 6; ++o) po[o] = (h16)pmax[o];
  }
  atomicMax(&sA, fkey(labs));
  atomicMax(&sM, fkey(lmax));
  __syncthreads();
  if (t == 0){ atomicMax(&red[1], sA); atomicMax(&red[2], sM); }
}

// ---------------- elementwise qrelu in place, 8 halves per thread ----------------
__global__ void qrelu_h8_kernel(h16* __restrict__ p, long n8, const float* s, int ia, int ib){
  long i = (long)blockIdx.x * blockDim.x + threadIdx.x;
  if (i < n8){
    v8h* vp = (v8h*)p + i;
    v8h v = *vp;
    float sa = s[ia], sb = s[ib];
    #pragma unroll
    for (int j = 0; j < 8; ++j) v[j] = (h16)qrelu_f((float)v[j], sa, sb);
    *vp = v;
  }
}

// ---------------- conv2 as WMMA implicit GEMM over NHWC activations ----------------
// M = B*100 pixels, N = 16 channels, K = dy*32 + dx*6 + ci (Kpad=160, row pad 30->32)
// k-step kk == tap row dy: lane's A fragment = two contiguous 16B runs (dword-aligned)
__global__ void __launch_bounds__(256) conv2_wmma_kernel(
    const h16* __restrict__ a1, const h16* __restrict__ w2p,
    const float* __restrict__ bias2, h16* __restrict__ y2, unsigned* red)
{
  int lane   = threadIdx.x & 31;
  int laneHi = lane >> 4;
  int ln     = lane & 15;
  int mtile  = (blockIdx.x * 256 + threadIdx.x) >> 5;

  // B fragments: contiguous 32B per lane per k-step -> vector loads
  v16h bfrag[5];
  #pragma unroll
  for (int kk = 0; kk < 5; ++kk)
    bfrag[kk] = *(const v16h*)(w2p + ln*160 + kk*32 + laneHi*16);

  int m  = mtile*16 + ln;            // output pixel of this lane's A row
  int b  = m / 100; int pix = m % 100;
  int oy = pix / 10, ox = pix % 10;
  int base_e = ((b*14 + oy)*14 + ox)*6;   // NHWC element index (fits i32)

  v8f c = {};
  #pragma unroll
  for (int kk = 0; kk < 5; ++kk){
    // row base for tap row dy=kk; lane's halves at +laneHi*16 (16 contiguous elems)
    const uint* p32 = (const uint*)(a1 + base_e + kk*84 + laneHi*16); // 4B-aligned
    union { v16h h; uint u[8]; } A;
    #pragma unroll
    for (int j = 0; j < 4; ++j) A.u[j]     = p32[j];      // K offsets laneHi*16 + 0..7
    #pragma unroll
    for (int j = 0; j < 4; ++j) A.u[4 + j] = p32[4 + j];  // K offsets laneHi*16 + 8..15
    c = __builtin_amdgcn_wmma_f32_16x16x32_f16(false, A.h, false, bfrag[kk],
                                               (short)0, c, false, false);
  }

  float bi = bias2[ln];
  float labs = 0.f, lmax = NEGBIG;
  #pragma unroll
  for (int r = 0; r < 8; ++r){       // C element r -> row r + 8*laneHi, col ln
    int pm = mtile*16 + r + laneHi*8;
    float y = c[r] + bi;
    labs = fmaxf(labs, fabsf(y));
    lmax = fmaxf(lmax, y);
    int bb = pm / 100; int pr = pm % 100;
    y2[((long)bb*16 + ln)*100 + pr] = (h16)y;   // [B,16,10,10]
  }
  #pragma unroll
  for (int msk = 16; msk > 0; msk >>= 1){
    labs = fmaxf(labs, __shfl_xor(labs, msk, 32));
    lmax = fmaxf(lmax, __shfl_xor(lmax, msk, 32));
  }
  if (lane == 0){ atomicMax(&red[3], fkey(labs)); atomicMax(&red[4], fkey(lmax)); }
}

// ---------------- pool 2x2 + qrelu -> flattened fp16 [B,416] (K-padded) ----------------
__global__ void pool_qrelu2_kernel(const h16* __restrict__ y2, h16* __restrict__ a2,
                                   const float* __restrict__ s){
  long idx = (long)blockIdx.x * blockDim.x + threadIdx.x;
  if (idx >= (long)BATCH * 416) return;
  int b = (int)(idx / 416); int f = (int)(idx % 416);
  if (f >= 400){ a2[idx] = (h16)0.f; return; }
  int ch = f / 25; int t = f % 25;
  int py = t / 5, px = t % 5;
  long base = ((long)b*16 + ch)*100 + py*20 + px*2;
  float v = fmaxf(fmaxf((float)y2[base],    (float)y2[base+1]),
                  fmaxf((float)y2[base+10], (float)y2[base+11]));
  a2[idx] = (h16)qrelu_f(v, s[3], s[4]);
}

// ---------------- generic WMMA FC: one wave = 16 rows x all N tiles ----------------
template<int NT, bool RED>
__global__ void __launch_bounds__(256) fc_wmma_kernel(
    const h16* __restrict__ A, const h16* __restrict__ W,
    const float* __restrict__ bias, float* __restrict__ out,
    int Kpad, int Nvalid, unsigned* redA, unsigned* redM)
{
  int lane   = threadIdx.x & 31;
  int laneHi = lane >> 4;
  int ln     = lane & 15;
  int wave   = (blockIdx.x * 256 + threadIdx.x) >> 5;
  int m0     = wave * 16;

  v8f c[NT] = {};
  int ksteps = Kpad >> 5;
  const h16* arow = A + (long)(m0 + ln) * Kpad;
  for (int kk = 0; kk < ksteps; ++kk){
    // A fragment = K{k0..k0+15} contiguous, k0 = kk*32 + laneHi*16
    const h16* ap = arow + kk*32 + laneHi*16;
    v8h alo = *(const v8h*)(ap);
    v8h ahi = *(const v8h*)(ap + 8);
    v16h a = __builtin_shufflevector(alo, ahi, 0,1,2,3,4,5,6,7,8,9,10,11,12,13,14,15);
    #pragma unroll
    for (int j = 0; j < NT; ++j){     // A fragment reused across all N tiles
      int n = j*16 + ln;
      v16h bf = *(const v16h*)(W + (long)n*Kpad + kk*32 + laneHi*16);  // contiguous 32B
      c[j] = __builtin_amdgcn_wmma_f32_16x16x32_f16(false, a, false, bf,
                                                    (short)0, c[j], false, false);
    }
  }

  float labs = 0.f, lmax = NEGBIG;
  #pragma unroll
  for (int j = 0; j < NT; ++j){
    int n = j*16 + ln;
    if (n < Nvalid){
      float bi = bias[n];
      #pragma unroll
      for (int r = 0; r < 8; ++r){
        float y = c[j][r] + bi;
        int mm = m0 + r + laneHi*8;
        out[(long)mm * Nvalid + n] = y;   // unpadded output width
        if (RED){ labs = fmaxf(labs, fabsf(y)); lmax = fmaxf(lmax, y); }
      }
    }
  }
  if (RED){
    #pragma unroll
    for (int msk = 16; msk > 0; msk >>= 1){
      labs = fmaxf(labs, __shfl_xor(labs, msk, 32));
      lmax = fmaxf(lmax, __shfl_xor(lmax, msk, 32));
    }
    if (lane == 0){ atomicMax(redA, fkey(labs)); atomicMax(redM, fkey(lmax)); }
  }
}

// ---------------- qrelu fp32 [B,Nvalid] -> fp16 [B,Npad] (zero-padded) ----------------
__global__ void qrelu_pad_kernel(const float* __restrict__ y, h16* __restrict__ a,
                                 int Nvalid, int Npad, const float* s, int ia, int ib){
  long idx = (long)blockIdx.x * blockDim.x + threadIdx.x;
  if (idx >= (long)BATCH * Npad) return;
  int m = (int)(idx / Npad); int n = (int)(idx % Npad);
  float v = 0.f;
  if (n < Nvalid) v = qrelu_f(y[(long)m*Nvalid + n], s[ia], s[ib]);
  a[idx] = (h16)v;
}

extern "C" void kernel_launch(void* const* d_in, const int* in_sizes, int n_in,
                              void* d_out, int out_size, void* d_ws, size_t ws_size,
                              hipStream_t stream)
{
  (void)in_sizes; (void)n_in; (void)out_size; (void)ws_size;
  const float* in  = (const float*)d_in[0];
  const float* w1  = (const float*)d_in[1];
  const float* b1  = (const float*)d_in[2];
  const float* w2  = (const float*)d_in[3];
  const float* b2  = (const float*)d_in[4];
  const float* fw1 = (const float*)d_in[5];
  const float* fb1 = (const float*)d_in[6];
  const float* fw2 = (const float*)d_in[7];
  const float* fb2 = (const float*)d_in[8];
  const float* fw3 = (const float*)d_in[9];
  const float* fb3 = (const float*)d_in[10];
  float* out = (float*)d_out;

  char* ws = (char*)d_ws;
  size_t off = 0;
  auto take = [&](size_t bytes)->char*{
    char* p = ws + off;
    off = (off + bytes + 255) & ~(size_t)255;
    return p;
  };
  unsigned* red  = (unsigned*)take(16*4);
  float*  svec   = (float*)take(16*4);
  float*  wq1    = (float*)take(450*4);
  float*  bias1  = (float*)take(6*4);
  h16*    w2p    = (h16*)take(16*160*2);
  float*  bias2  = (float*)take(16*4);
  h16*    f1p    = (h16*)take((size_t)128*416*2);
  float*  fb1p   = (float*)take(128*4);
  h16*    f2p    = (h16*)take((size_t)96*128*2);
  float*  fb2p   = (float*)take(96*4);
  h16*    f3p    = (h16*)take((size_t)16*96*2);
  float*  fb3p   = (float*)take(16*4);
  h16*    p1     = (h16*)take((size_t)BATCH*1176*2 + 256); // NHWC pooled conv1 pre-act (+row-pad over-read slack)
  h16*    y2     = (h16*)take((size_t)BATCH*1600*2);       // conv2 pre-act [B,16,10,10]
  h16*    a2     = (h16*)take((size_t)BATCH*416*2);        // flattened, K-padded
  float*  y3     = (float*)take((size_t)BATCH*120*4);
  h16*    a3     = (h16*)take((size_t)BATCH*128*2);
  float*  y4     = (float*)take((size_t)BATCH*84*4);
  h16*    a4     = (h16*)take((size_t)BATCH*96*2);

  init_kernel<<<1, 64, 0, stream>>>(red, svec);
  prep_weights_kernel<<<5, 256, 0, stream>>>(w1,b1,w2,b2,fw1,fb1,fw2,fb2,fw3,fb3,
      wq1,bias1,w2p,bias2,f1p,fb1p,f2p,fb2p,f3p,fb3p);

  absmax4_kernel<<<4096, 256, 0, stream>>>((const float4*)in, (long)BATCH*3072/4, red);
  scale0_kernel<<<1, 32, 0, stream>>>(red, svec);

  conv1_pool_kernel<<<(BATCH*196)/256, 256, 0, stream>>>(in, wq1, bias1, svec, p1, red);
  scale2_kernel<<<1, 32, 0, stream>>>(red, svec, 1, 2, 1, 2);
  qrelu_h8_kernel<<<(unsigned)(((long)BATCH*1176/8 + 255)/256), 256, 0, stream>>>(
      p1, (long)BATCH*1176/8, svec, 1, 2);

  conv2_wmma_kernel<<<12800, 256, 0, stream>>>(p1, w2p, bias2, y2, red); // 102400 m-tiles
  scale2_kernel<<<1, 32, 0, stream>>>(red, svec, 3, 4, 3, 4);
  pool_qrelu2_kernel<<<(BATCH*416)/256, 256, 0, stream>>>(y2, a2, svec);

  fc_wmma_kernel<8, true><<<128, 256, 0, stream>>>(a2, f1p, fb1p, y3, 416, 120, &red[5], &red[6]);
  scale2_kernel<<<1, 32, 0, stream>>>(red, svec, 5, 6, 5, 6);
  qrelu_pad_kernel<<<(BATCH*128)/256, 256, 0, stream>>>(y3, a3, 120, 128, svec, 5, 6);

  fc_wmma_kernel<6, true><<<128, 256, 0, stream>>>(a3, f2p, fb2p, y4, 128, 84, &red[7], &red[8]);
  scale2_kernel<<<1, 32, 0, stream>>>(red, svec, 7, 8, 7, 8);
  qrelu_pad_kernel<<<(BATCH*96)/256, 256, 0, stream>>>(y4, a4, 84, 96, svec, 7, 8);

  fc_wmma_kernel<1, false><<<128, 256, 0, stream>>>(a4, f3p, fb3p, out, 96, 10, nullptr, nullptr);
}
